// EntireGNN_60748017434772
// MI455X (gfx1250) — matrix operand order repaired and checked by reading, not compiled, not run on hardware
//
#include <hip/hip_runtime.h>
#include <hip/hip_bf16.h>

#define GG 8
#define NN 20000
#define EE 320000
#define STATE 64
#define HID 128
#define EMBD 64
#define POSN 16
#define LDOUT 128   // row stride for tmp/out buffers (fixed for all layers)

typedef __bf16 bf16_t;
typedef bf16_t v16bf __attribute__((ext_vector_type(16)));
typedef bf16_t bf16x2 __attribute__((ext_vector_type(2)));
typedef float  v8f   __attribute__((ext_vector_type(8)));
typedef unsigned int v8ui __attribute__((ext_vector_type(8)));

// pack two f32 into one dword of bf16 pair using native converts
__device__ __forceinline__ unsigned pack_bf16x2(float lo, float hi) {
    bf16x2 p = { (bf16_t)lo, (bf16_t)hi };
    return __builtin_bit_cast(unsigned, p);
}

// ---------------- degree / dinv ----------------
__global__ void deg_init(float* deg, int n) {
    int i = blockIdx.x * blockDim.x + threadIdx.x;
    if (i < n) deg[i] = 1.0f;  // self-loop weight
}

__global__ void deg_accum(const int* __restrict__ ei, const float* __restrict__ ew,
                          float* __restrict__ deg) {
    int e = blockIdx.x * blockDim.x + threadIdx.x;
    int g = blockIdx.y;
    if (e >= EE) return;
    int dst = ei[(size_t)g * 2 * EE + EE + e];
    atomicAdd(&deg[(size_t)g * NN + dst], ew[(size_t)g * EE + e]);
}

__global__ void deg_finish(float* deg, int n) {
    int i = blockIdx.x * blockDim.x + threadIdx.x;
    if (i < n) deg[i] = rsqrtf(deg[i]);   // deg buffer now holds dinv
}

// ---------------- WMMA GEMM: tmp = A @ W ; out = dinv^2 * tmp + b ----------------
// A: [G,N,K] row-major (stride == K). tmp/out: [G,N,LDOUT], cols [0,OUT) written.
template<int K, int OUT, bool RELU>
__global__ __launch_bounds__(256)
void gcn_gemm(const float* __restrict__ A, const float* __restrict__ W,
              const float* __restrict__ bias, const float* __restrict__ dinv,
              float* __restrict__ tmp, float* __restrict__ outp)
{
    // Fragment-major staged weights: wlds[col][kpair], padded stride to rotate
    // LDS banks by 4 per column -> conflict-free ds_load_b128 B-fragment reads.
    constexpr int KP  = K / 2;        // bf16 pairs along K
    constexpr int LDW = KP + 4;       // padded stride (dwords)
    __shared__ unsigned wlds[OUT * LDW];

    const int tid = threadIdx.x;
    const int g   = blockIdx.y;
    constexpr int NCT = OUT / 16;              // col tiles
    constexpr int RPB = 16 * (8 / NCT);        // rows per block (8 waves per block)

    for (int idx = tid; idx < OUT * KP; idx += 256) {
        int c = idx / KP, kp = idx % KP;
        wlds[c * LDW + kp] = pack_bf16x2(W[(2 * kp) * OUT + c],
                                         W[(2 * kp + 1) * OUT + c]);
    }
    __syncthreads();

    const int wave = tid >> 5;
    const int lane = tid & 31;
    const int hi16 = lane >> 4;
    const int l16  = lane & 15;
    const int ct   = wave % NCT;
    const int rt   = wave / NCT;
    const int rowbase = blockIdx.x * RPB + rt * 16;
    const int colbase = ct * 16;
    const size_t abase = ((size_t)g * NN + (rowbase + l16)) * (size_t)K;
    const unsigned* wcol = wlds + (colbase + l16) * LDW;  // this lane's B column

    v8f acc = {};
    #pragma unroll
    for (int kb = 0; kb < K; kb += 32) {
        // A fragment (ISA 7.12.2 16-bit A): lane<16 -> row=rowbase+l16,
        // K = {kb..kb+7, kb+16..kb+23}; lane>=16 shifts K base by 8.
        const float4* ap = (const float4*)(A + abase + kb + hi16 * 8);
        float4 a0 = ap[0], a1 = ap[1];   // K +0..7
        float4 a2 = ap[4], a3 = ap[5];   // K +16..23
        if (RELU) {
            a0.x=fmaxf(a0.x,0.f); a0.y=fmaxf(a0.y,0.f); a0.z=fmaxf(a0.z,0.f); a0.w=fmaxf(a0.w,0.f);
            a1.x=fmaxf(a1.x,0.f); a1.y=fmaxf(a1.y,0.f); a1.z=fmaxf(a1.z,0.f); a1.w=fmaxf(a1.w,0.f);
            a2.x=fmaxf(a2.x,0.f); a2.y=fmaxf(a2.y,0.f); a2.z=fmaxf(a2.z,0.f); a2.w=fmaxf(a2.w,0.f);
            a3.x=fmaxf(a3.x,0.f); a3.y=fmaxf(a3.y,0.f); a3.z=fmaxf(a3.z,0.f); a3.w=fmaxf(a3.w,0.f);
        }
        v16bf afrag = {
            (bf16_t)a0.x, (bf16_t)a0.y, (bf16_t)a0.z, (bf16_t)a0.w,
            (bf16_t)a1.x, (bf16_t)a1.y, (bf16_t)a1.z, (bf16_t)a1.w,
            (bf16_t)a2.x, (bf16_t)a2.y, (bf16_t)a2.z, (bf16_t)a2.w,
            (bf16_t)a3.x, (bf16_t)a3.y, (bf16_t)a3.z, (bf16_t)a3.w };

        // B fragment: col=colbase+l16, K = {kb..kb+7, kb+16..kb+23};
        // lanes 16-31 shift K base by 8 (matches A fragment).
        const unsigned* bp = wcol + ((kb + hi16 * 8) >> 1);
        v8ui braw;
        #pragma unroll
        for (int j = 0; j < 8; ++j) braw[j] = bp[(j & 3) + (j >> 2) * 8];
        v16bf bfrag = __builtin_bit_cast(v16bf, braw);

        acc = __builtin_amdgcn_wmma_f32_16x16x32_bf16(
                  false, afrag, false, bfrag, (short)0, acc, false, false);
    }

    // In-place safety: every A row read by this block is written only by this
    // block; barrier guarantees all reads precede any write.
    __syncthreads();

    const int col = colbase + l16;
    const float bv = bias[col];
    #pragma unroll
    for (int v = 0; v < 8; ++v) {
        int row = rowbase + hi16 * 8 + v;           // C/D layout: VGPR v <-> row
        size_t o = ((size_t)g * NN + row) * LDOUT + col;
        float val = acc[v];
        tmp[o] = val;                                // raw xW for edge scatter
        float di = dinv[(size_t)g * NN + row];
        outp[o] = val * di * di + bv;                // self-loop term + bias
    }
}

// ---------------- edge scatter: out[dst] += dinv[src]*w*dinv[dst] * tmp[src] ----------------
template<int OUT>
__global__ __launch_bounds__(256)
void gcn_scatter(const int* __restrict__ ei, const float* __restrict__ ew,
                 const float* __restrict__ dinv, const float* __restrict__ tmp,
                 float* __restrict__ outp)
{
    constexpr int CH = OUT / 4;                     // float4 chunks per edge
    const int g = blockIdx.y;
    long long t = (long long)blockIdx.x * blockDim.x + threadIdx.x;
    long long e = t / CH;
    int ch = (int)(t % CH);
    if (e >= EE) return;
    const int* eg = ei + (size_t)g * 2 * EE;
    int src = eg[e];
    int dst = eg[EE + e];
    float w = ew[(size_t)g * EE + e];
    float norm = dinv[(size_t)g * NN + src] * w * dinv[(size_t)g * NN + dst];
    float4 v = ((const float4*)(tmp + ((size_t)g * NN + src) * LDOUT))[ch];
    float* o = outp + ((size_t)g * NN + dst) * LDOUT + ch * 4;
    atomicAdd(o + 0, norm * v.x);
    atomicAdd(o + 1, norm * v.y);
    atomicAdd(o + 2, norm * v.z);
    atomicAdd(o + 3, norm * v.w);
}

// ---------------- final gather ----------------
__global__ void gather_out(const float* __restrict__ emb, const int* __restrict__ pos,
                           float* __restrict__ outp)
{
    int idx = blockIdx.x * blockDim.x + threadIdx.x;
    if (idx >= GG * POSN * EMBD) return;
    int c = idx % EMBD;
    int p = (idx / EMBD) % POSN;
    int g = idx / (EMBD * POSN);
    int pp = pos[g * POSN + p];
    int safe = pp > 0 ? pp : 0;
    float v = emb[((size_t)g * NN + safe) * LDOUT + c];
    outp[idx] = (pp != -1) ? v : -4.0f;   // -DEPTH
}

extern "C" void kernel_launch(void* const* d_in, const int* in_sizes, int n_in,
                              void* d_out, int out_size, void* d_ws, size_t ws_size,
                              hipStream_t stream) {
    (void)in_sizes; (void)n_in; (void)out_size; (void)ws_size;
    const float* x   = (const float*)d_in[0];
    const int*   ei  = (const int*)d_in[1];
    const float* ew  = (const float*)d_in[2];
    const int*   pos = (const int*)d_in[3];
    const float* W0  = (const float*)d_in[4];
    const float* b0  = (const float*)d_in[5];
    const float* W1  = (const float*)d_in[6];
    const float* b1  = (const float*)d_in[7];
    const float* W2  = (const float*)d_in[8];
    const float* b2  = (const float*)d_in[9];
    float* out = (float*)d_out;

    // workspace: tmp[G,N,128] | buf[G,N,128] | dinv[G,N]  (~165 MB)
    float* tmp  = (float*)d_ws;
    float* buf  = tmp + (size_t)GG * NN * LDOUT;
    float* dinv = buf + (size_t)GG * NN * LDOUT;

    // degrees -> dinv
    deg_init<<<(GG * NN + 255) / 256, 256, 0, stream>>>(dinv, GG * NN);
    {
        dim3 grid((EE + 255) / 256, GG);
        deg_accum<<<grid, 256, 0, stream>>>(ei, ew, dinv);
    }
    deg_finish<<<(GG * NN + 255) / 256, 256, 0, stream>>>(dinv, GG * NN);

    // Layer 0: K=64 (stride 64 input x), OUT=128, no relu on input
    {
        dim3 grid(NN / 16, GG);
        gcn_gemm<64, 128, false><<<grid, 256, 0, stream>>>(x, W0, b0, dinv, tmp, buf);
        dim3 sg((long long)EE * (128 / 4) / 256 + 1, GG);
        gcn_scatter<128><<<sg, 256, 0, stream>>>(ei, ew, dinv, tmp, buf);
    }
    // Layer 1: K=128, OUT=128, relu on input, in-place on buf
    {
        dim3 grid(NN / 16, GG);
        gcn_gemm<128, 128, true><<<grid, 256, 0, stream>>>(buf, W1, b1, dinv, tmp, buf);
        dim3 sg((long long)EE * (128 / 4) / 256 + 1, GG);
        gcn_scatter<128><<<sg, 256, 0, stream>>>(ei, ew, dinv, tmp, buf);
    }
    // Layer 2: K=128, OUT=64, relu on input, in-place on buf (row stride stays 128)
    {
        dim3 grid(NN / 32, GG);
        gcn_gemm<128, 64, true><<<grid, 256, 0, stream>>>(buf, W2, b2, dinv, tmp, buf);
        dim3 sg((long long)EE * (64 / 4) / 256 + 1, GG);
        gcn_scatter<64><<<sg, 256, 0, stream>>>(ei, ew, dinv, tmp, buf);
    }
    // Gather 16 positions per graph from pre-activation layer-2 output
    gather_out<<<(GG * POSN * EMBD + 255) / 256, 256, 0, stream>>>(buf, pos, out);
}
